// SpecialFC_2516850835703
// MI455X (gfx1250) — matrix operand order repaired and checked by reading, not compiled, not run on hardware
//
#include <hip/hip_runtime.h>

typedef __attribute__((ext_vector_type(16))) __bf16 v16bf;
typedef __attribute__((ext_vector_type(8)))  float  v8f;

namespace {
constexpr int kB    = 8;
constexpr int kN    = 8192;
constexpr int kCin  = 512;
constexpr int kCout = 512;
constexpr int TILE_N = 128;          // output rows per workgroup
constexpr int TILE_O = 128;          // output cols per workgroup
constexpr int KSTEP  = 32;           // bf16 WMMA K depth
constexpr int NKT    = kCin / KSTEP; // 16 K-steps
constexpr int LROW   = 20;           // dwords per LDS row: 16 data + 4 pad (80B stride -> conflict-free)
}

union FragU {
    uint4 q[2];
    v16bf v;
};

// 256 threads = 8 wave32 waves, tiled 4 (rows) x 2 (cols).
// Each wave computes 32(n) x 64(o) = 2x4 grid of 16x16 WMMA accumulators.
__global__ __launch_bounds__(256)
void batched_linear_wmma_bf16(const float* __restrict__ X,
                              const float* __restrict__ W,
                              const float* __restrict__ bias,
                              float* __restrict__ out)
{
    __shared__ unsigned int sA[2][TILE_N * LROW];   // x tile, bf16 packed in dwords
    __shared__ unsigned int sW[2][TILE_O * LROW];   // W tile, bf16 packed in dwords

    const int b  = blockIdx.z;
    const int n0 = blockIdx.y * TILE_N;
    const int o0 = blockIdx.x * TILE_O;

    const int tid  = threadIdx.x;
    const int lane = tid & 31;
    const int wave = tid >> 5;
    const int wrow = wave & 3;    // 0..3 -> 32-row slab
    const int wcol = wave >> 2;   // 0..1 -> 64-col slab
    const int lm   = lane & 15;   // 0..15
    const int lhi  = lane >> 4;   // 0 or 1

    const float* Xb = X + (size_t)b * kN * kCin;
    const float* Wb = W + (size_t)b * kCout * kCin;

    // ---- cooperative staging: each thread owns 16 contiguous floats of one tile row
    const int srow = tid >> 1;          // 0..127
    const int scol = (tid & 1) * 16;    // 0 or 16 (float / bf16 column)
    const float* gx = Xb + (size_t)(n0 + srow) * kCin + scol;
    const float* gw = Wb + (size_t)(o0 + srow) * kCin + scol;
    const int soff = srow * LROW + (scol >> 1);   // dword offset into LDS tile

    float4 xl[4], wl[4];

    auto gload = [&](int k0) {
#pragma unroll
        for (int i = 0; i < 4; ++i) {
            xl[i] = *(const float4*)(gx + k0 + 4 * i);
            wl[i] = *(const float4*)(gw + k0 + 4 * i);
        }
    };

    auto sstore = [&](int buf) {
        union { __bf16 h[16]; uint4 q[2]; } ux, uw;
#pragma unroll
        for (int i = 0; i < 4; ++i) {
            ux.h[4 * i + 0] = (__bf16)xl[i].x;  ux.h[4 * i + 1] = (__bf16)xl[i].y;
            ux.h[4 * i + 2] = (__bf16)xl[i].z;  ux.h[4 * i + 3] = (__bf16)xl[i].w;
            uw.h[4 * i + 0] = (__bf16)wl[i].x;  uw.h[4 * i + 1] = (__bf16)wl[i].y;
            uw.h[4 * i + 2] = (__bf16)wl[i].z;  uw.h[4 * i + 3] = (__bf16)wl[i].w;
        }
        uint4* pa = (uint4*)(&sA[buf][soff]);
        uint4* pw = (uint4*)(&sW[buf][soff]);
        pa[0] = ux.q[0];  pa[1] = ux.q[1];
        pw[0] = uw.q[0];  pw[1] = uw.q[1];
    };

    v8f acc[2][4] = {};

    // prologue: fill buffer 0
    gload(0);
    sstore(0);
    __syncthreads();

    for (int kt = 0; kt < NKT; ++kt) {
        const int cur = kt & 1;

        // issue next tile's global loads early; latency hides under WMMA work
        if (kt + 1 < NKT) gload((kt + 1) * KSTEP);

        const unsigned int* Ab = sA[cur];
        const unsigned int* Wt = sW[cur];

        // ---- A fragments: lane l holds row (l&15), K = (l>>4)*8 + {0..7, 16..23}
        v16bf a[2];
#pragma unroll
        for (int r = 0; r < 2; ++r) {
            const int row = wrow * 32 + r * 16 + lm;
            FragU f;
            f.q[0] = *(const uint4*)(Ab + row * LROW + lhi * 4);
            f.q[1] = *(const uint4*)(Ab + row * LROW + lhi * 4 + 8);
            a[r] = f.v;
        }
        // ---- B fragments: lane l holds col (l&15), K = (l>>4)*8 + {0..7, 16..23}
        v16bf bm[4];
#pragma unroll
        for (int c = 0; c < 4; ++c) {
            const int row = wcol * 64 + c * 16 + lm;
            FragU f;
            f.q[0] = *(const uint4*)(Wt + row * LROW + lhi * 4);
            f.q[1] = *(const uint4*)(Wt + row * LROW + lhi * 4 + 8);
            bm[c] = f.v;
        }

#pragma unroll
        for (int r = 0; r < 2; ++r)
#pragma unroll
            for (int c = 0; c < 4; ++c)
                acc[r][c] = __builtin_amdgcn_wmma_f32_16x16x32_bf16(
                    false, a[r], false, bm[c], (short)0, acc[r][c], false, false);

        // convert + stage next tile into the other buffer
        if (kt + 1 < NKT) sstore(cur ^ 1);
        __syncthreads();
    }

    // ---- Epilogue: bias add + non-temporal f32 stores.
    // C/D layout: VGPR v <-> row (v + 8*(lane>>4)), col (lane&15).
#pragma unroll
    for (int c = 0; c < 4; ++c) {
        const float bv = bias[b * kCout + o0 + wcol * 64 + c * 16 + lm];
#pragma unroll
        for (int r = 0; r < 2; ++r) {
            float* orow = out
                + ((size_t)b * kN + n0 + wrow * 32 + r * 16 + lhi * 8) * kCout
                + o0 + wcol * 64 + c * 16 + lm;
#pragma unroll
            for (int v = 0; v < 8; ++v)
                __builtin_nontemporal_store(acc[r][c][v] + bv,
                                            &orow[(size_t)v * kCout]);
        }
    }
}

extern "C" void kernel_launch(void* const* d_in, const int* in_sizes, int n_in,
                              void* d_out, int out_size, void* d_ws, size_t ws_size,
                              hipStream_t stream) {
    const float* x  = (const float*)d_in[0];   // [B, N, C_IN]
    const float* W  = (const float*)d_in[1];   // [B, C_OUT, C_IN]
    const float* bs = (const float*)d_in[2];   // [B, C_OUT]
    float* out = (float*)d_out;                // [B, N, C_OUT]

    dim3 grid(kCout / TILE_O, kN / TILE_N, kB);  // (4, 64, 8)
    dim3 block(256);
    hipLaunchKernelGGL(batched_linear_wmma_bf16, grid, block, 0, stream,
                       x, W, bs, out);
}
